// MambaModel_42511586295927
// MI455X (gfx1250) — compile-verified, matrix operand order checked by reading, not run on hardware
//
#include <hip/hip_runtime.h>

// ---------------- model dims ----------------
#define D_MODEL   512
#define D_INNER   1024
#define D_STATE   16
#define D_CONV    4
#define DT_RANK   32
#define NUM_LAYERS 6
#define BATCH     8
#define SEQ       512
#define IN_DIM    16
#define OUT_DIM   6
#define MTOT      (BATCH * SEQ)        // 4096 rows for all token-parallel GEMMs

typedef __attribute__((ext_vector_type(16))) __bf16 v16bf;
typedef __attribute__((ext_vector_type(8)))  float  v8f;

union FragBF {
    uint4  u[2];   // 2 x 16B global loads
    v16bf  v;      // WMMA operand
};

// round-to-nearest-even f32 -> bf16 (stored as ushort)
__device__ __forceinline__ unsigned short f2bf(float f) {
    union { float f; unsigned u; } v; v.f = f;
    unsigned r = v.u + 0x7FFFu + ((v.u >> 16) & 1u);
    return (unsigned short)(r >> 16);
}

__device__ __forceinline__ float silu(float x) {
    return x / (1.0f + __expf(-x));
}

// ---------------------------------------------------------------------------
// Register-blocked bf16 WMMA GEMM:
//   C[M,N](f32) = A[M,K](bf16,row lda) * W[N,K](bf16,row ldw)^T
// Each wave computes a 32x64 slab as 2(M) x 4(N) 16x16 tiles: per k-step of 32
// it loads 2 A fragments + 4 B fragments (6 x 32B/lane) and issues 8
// v_wmma_f32_16x16x32_bf16 -> 4x A reuse, 2x B reuse, ~21 flop/B from L2.
// 4 waves/block stack along M (block tile 128x64); their identical B-fragment
// addresses are served by the WGP cache broadcast. Grid: (N/64, M/128).
// ---------------------------------------------------------------------------
__global__ void __launch_bounds__(128)
wmma_gemm_bf16(const unsigned short* __restrict__ A, int lda,
               const unsigned short* __restrict__ W, int ldw,
               float* __restrict__ C, int ldc,
               int M, int N, int K)
{
    const int lane = threadIdx.x & 31;
    const int wave = threadIdx.x >> 5;
    const int tm = (blockIdx.y * 4 + wave) * 32;   // 2 row tiles per wave
    const int tn = blockIdx.x * 64;                // 4 col tiles per wave
    if (tm >= M || tn >= N) return;

    const int r     = lane & 15;            // row within tile (A) / W row (B)
    const int khalf = (lane >> 4) * 8;      // K sub-chunk base per ISA layout

    const unsigned short* a0 = A + (size_t)(tm + r) * lda + khalf;
    const unsigned short* a1 = a0 + (size_t)16 * lda;
    const unsigned short* b0 = W + (size_t)(tn + r) * ldw + khalf;

    v8f acc[2][4] = {};
    for (int k0 = 0; k0 < K; k0 += 32) {
        FragBF a[2], b[4];
        a[0].u[0] = *(const uint4*)(a0 + k0);
        a[0].u[1] = *(const uint4*)(a0 + k0 + 16);
        a[1].u[0] = *(const uint4*)(a1 + k0);
        a[1].u[1] = *(const uint4*)(a1 + k0 + 16);
#pragma unroll
        for (int ni = 0; ni < 4; ++ni) {
            const unsigned short* bp = b0 + (size_t)ni * 16 * ldw + k0;
            b[ni].u[0] = *(const uint4*)(bp);
            b[ni].u[1] = *(const uint4*)(bp + 16);
        }
        if (k0 + 32 < K) {   // keep next fragments warm (global_prefetch_b8)
            __builtin_prefetch(a0 + k0 + 32, 0, 1);
            __builtin_prefetch(b0 + k0 + 32, 0, 1);
        }
#pragma unroll
        for (int mi = 0; mi < 2; ++mi)
#pragma unroll
            for (int ni = 0; ni < 4; ++ni)
                acc[mi][ni] = __builtin_amdgcn_wmma_f32_16x16x32_bf16(
                    false, a[mi].v, false, b[ni].v, (short)0, acc[mi][ni],
                    false, false);
    }

    // D layout: lane L, vgpr i -> row = i + 8*(L>=16), col = L&15
    const int mbase = (lane >> 4) << 3;
    const int nbase = lane & 15;
#pragma unroll
    for (int mi = 0; mi < 2; ++mi) {
        float* cr = C + (size_t)(tm + mi * 16 + mbase) * ldc + tn + nbase;
#pragma unroll
        for (int ni = 0; ni < 4; ++ni)
#pragma unroll
            for (int i = 0; i < 8; ++i)
                cr[(size_t)i * ldc + ni * 16] = acc[mi][ni][i];
    }
}

// ---------------------------------------------------------------------------
// f32 -> bf16 cast, 4 elements/thread (all sizes are multiples of 4)
// ---------------------------------------------------------------------------
__global__ void cast_bf16_kernel(const float* __restrict__ src,
                                 unsigned short* __restrict__ dst, int n4)
{
    int i = blockIdx.x * blockDim.x + threadIdx.x;
    if (i < n4) {
        float4 f = ((const float4*)src)[i];
        ushort4 o;
        o.x = f2bf(f.x); o.y = f2bf(f.y); o.z = f2bf(f.z); o.w = f2bf(f.w);
        ((ushort4*)dst)[i] = o;
    }
}

// ---------------------------------------------------------------------------
// Embedding: h[m][j] = sum_k x[m][k] * emb_w[j][k] + emb_b[j]   (K=16, trivial)
// ---------------------------------------------------------------------------
__global__ void embed_kernel(const float* __restrict__ x,
                             const float* __restrict__ emb_w,
                             const float* __restrict__ emb_b,
                             float* __restrict__ h,
                             unsigned short* __restrict__ hbf)
{
    int idx = blockIdx.x * blockDim.x + threadIdx.x;   // MTOT*D_MODEL threads
    int m = idx >> 9, j = idx & 511;
    const float* xr = x + (size_t)m * IN_DIM;
    const float* wr = emb_w + (size_t)j * IN_DIM;
    float acc = emb_b[j];
#pragma unroll
    for (int k = 0; k < IN_DIM; ++k) acc += xr[k] * wr[k];
    h[idx] = acc;
    hbf[idx] = f2bf(acc);
}

// ---------------------------------------------------------------------------
// Causal depthwise conv (width 4) + bias + SiLU, from xz's first half.
// Writes xi in f32 (for scan) and bf16 (for x_proj GEMM).
// ---------------------------------------------------------------------------
__global__ void conv_silu_kernel(const float* __restrict__ xz,
                                 const float* __restrict__ cw,  // [D_INNER,4]
                                 const float* __restrict__ cb,  // [D_INNER]
                                 float* __restrict__ xi,
                                 unsigned short* __restrict__ xibf)
{
    int idx = blockIdx.x * blockDim.x + threadIdx.x;   // BATCH*SEQ*D_INNER
    int d = idx & (D_INNER - 1);
    int t = (idx >> 10) & (SEQ - 1);
    int b = idx >> 19;
    const float* w = cw + d * D_CONV;
    float acc = cb[d];
#pragma unroll
    for (int j = 0; j < D_CONV; ++j) {
        int tt = t - (D_CONV - 1) + j;
        if (tt >= 0)
            acc += xz[((size_t)(b * SEQ + tt)) * (2 * D_INNER) + d] * w[j];
    }
    float v = silu(acc);
    xi[idx] = v;
    xibf[idx] = f2bf(v);
}

// ---------------------------------------------------------------------------
// Selective scan: sequential over t, parallel over (batch, channel).
// Fuses: dt bias + softplus, exp recurrence, C contraction, D skip,
// SiLU(z) gate, bf16 quantization for the out_proj GEMM.
// B_t / C_t (shared by all channels of a token) staged in LDS.
// ---------------------------------------------------------------------------
__global__ void __launch_bounds__(256)
scan_kernel(const float* __restrict__ dtproj,  // [MTOT, D_INNER]
            const float* __restrict__ dtb,     // [D_INNER]
            const float* __restrict__ dbc,     // [MTOT, 64]
            const float* __restrict__ xi,      // [MTOT, D_INNER]
            const float* __restrict__ A_log,   // [D_INNER, D_STATE]
            const float* __restrict__ Dp,      // [D_INNER]
            const float* __restrict__ xz,      // [MTOT, 2*D_INNER] (z half)
            unsigned short* __restrict__ ybf)  // [MTOT, D_INNER]
{
    const int b = blockIdx.y;
    const int d = blockIdx.x * blockDim.x + threadIdx.x;

    float A[D_STATE], h[D_STATE];
#pragma unroll
    for (int n = 0; n < D_STATE; ++n) {
        A[n] = -__expf(A_log[(size_t)d * D_STATE + n]);
        h[n] = 0.0f;
    }
    const float dbias = dtb[d];
    const float Dd    = Dp[d];

    __shared__ float sB[D_STATE], sC[D_STATE];

    for (int t = 0; t < SEQ; ++t) {
        const size_t row = (size_t)(b * SEQ + t);
        if (threadIdx.x < 2 * D_STATE) {
            float v = dbc[row * (DT_RANK + 2 * D_STATE) + DT_RANK + threadIdx.x];
            if (threadIdx.x < D_STATE) sB[threadIdx.x] = v;
            else                       sC[threadIdx.x - D_STATE] = v;
        }
        __syncthreads();

        float dv = dtproj[row * D_INNER + d] + dbias;
        dv = (dv > 20.0f) ? dv : log1pf(__expf(dv));      // softplus
        const float xv = xi[row * D_INNER + d];
        const float dx = dv * xv;

        float y = 0.0f;
#pragma unroll
        for (int n = 0; n < D_STATE; ++n) {
            h[n] = __expf(dv * A[n]) * h[n] + dx * sB[n];
            y += h[n] * sC[n];
        }

        const float zv = xz[row * (2 * D_INNER) + D_INNER + d];
        ybf[row * D_INNER + d] = f2bf((y + xv * Dd) * silu(zv));
        __syncthreads();
    }
}

// ---------------------------------------------------------------------------
// Head: out[b][j] = h[b, L-1, :] . head_w[j] + head_b[j]   (8x6, trivial)
// ---------------------------------------------------------------------------
__global__ void head_kernel(const float* __restrict__ h,
                            const float* __restrict__ hw,
                            const float* __restrict__ hb,
                            float* __restrict__ out)
{
    int i = threadIdx.x;
    if (i >= BATCH * OUT_DIM) return;
    int b = i / OUT_DIM, j = i % OUT_DIM;
    const float* hr = h + ((size_t)(b * SEQ + SEQ - 1)) * D_MODEL;
    const float* wr = hw + (size_t)j * D_MODEL;
    float acc = hb[j];
    for (int k = 0; k < D_MODEL; ++k) acc += hr[k] * wr[k];
    out[i] = acc;
}

// ---------------------------------------------------------------------------

extern "C" void kernel_launch(void* const* d_in, const int* in_sizes, int n_in,
                              void* d_out, int out_size, void* d_ws, size_t ws_size,
                              hipStream_t stream) {
    const float* x       = (const float*)d_in[0];
    const float* emb_w   = (const float*)d_in[1];
    const float* emb_b   = (const float*)d_in[2];
    const float* in_w    = (const float*)d_in[3];
    const float* conv_w  = (const float*)d_in[4];
    const float* conv_b  = (const float*)d_in[5];
    const float* xproj_w = (const float*)d_in[6];
    const float* dt_w    = (const float*)d_in[7];
    const float* dt_b    = (const float*)d_in[8];
    const float* A_log   = (const float*)d_in[9];
    const float* Dparam  = (const float*)d_in[10];
    const float* out_w   = (const float*)d_in[11];
    const float* head_w  = (const float*)d_in[12];
    const float* head_b  = (const float*)d_in[13];
    float* out = (float*)d_out;

    // -------- workspace carve-up --------
    char* ws = (char*)d_ws;
    size_t off = 0;
    auto carve = [&](size_t bytes) -> void* {
        void* p = ws + off;
        off = (off + bytes + 255) & ~(size_t)255;
        return p;
    };
    float*          h_f32   = (float*)         carve((size_t)MTOT * D_MODEL * 4);
    unsigned short* h_bf    = (unsigned short*)carve((size_t)MTOT * D_MODEL * 2);
    float*          xz      = (float*)         carve((size_t)MTOT * 2 * D_INNER * 4);
    float*          xi_f32  = (float*)         carve((size_t)MTOT * D_INNER * 4);
    unsigned short* xi_bf   = (unsigned short*)carve((size_t)MTOT * D_INNER * 2);
    float*          dbc     = (float*)         carve((size_t)MTOT * 64 * 4);
    unsigned short* dbc_bf  = (unsigned short*)carve((size_t)MTOT * 64 * 2);
    float*          dtproj  = (float*)         carve((size_t)MTOT * D_INNER * 4);
    unsigned short* y_bf    = (unsigned short*)carve((size_t)MTOT * D_INNER * 2);
    unsigned short* w_in_bf = (unsigned short*)carve((size_t)NUM_LAYERS * 2 * D_INNER * D_MODEL * 2);
    unsigned short* w_xp_bf = (unsigned short*)carve((size_t)NUM_LAYERS * 64 * D_INNER * 2);
    unsigned short* w_dt_bf = (unsigned short*)carve((size_t)NUM_LAYERS * D_INNER * DT_RANK * 2);
    unsigned short* w_out_bf= (unsigned short*)carve((size_t)NUM_LAYERS * D_MODEL * D_INNER * 2);

    auto castN = [&](const float* s, unsigned short* dst, int n) {
        int n4 = n >> 2;
        cast_bf16_kernel<<<(n4 + 255) / 256, 256, 0, stream>>>(s, dst, n4);
    };

    // -------- weights -> bf16 (deterministic, every call) --------
    castN(in_w,    w_in_bf,  NUM_LAYERS * 2 * D_INNER * D_MODEL);
    castN(xproj_w, w_xp_bf,  NUM_LAYERS * 64 * D_INNER);
    castN(dt_w,    w_dt_bf,  NUM_LAYERS * D_INNER * DT_RANK);
    castN(out_w,   w_out_bf, NUM_LAYERS * D_MODEL * D_INNER);

    // -------- embedding --------
    embed_kernel<<<(MTOT * D_MODEL) / 256, 256, 0, stream>>>(x, emb_w, emb_b, h_f32, h_bf);

    // -------- layers --------
    for (int i = 0; i < NUM_LAYERS; ++i) {
        // xz = h @ in_w^T : M=4096, N=2048, K=512
        wmma_gemm_bf16<<<dim3(2 * D_INNER / 64, MTOT / 128), 128, 0, stream>>>(
            h_bf, D_MODEL,
            w_in_bf + (size_t)i * 2 * D_INNER * D_MODEL, D_MODEL,
            xz, 2 * D_INNER, MTOT, 2 * D_INNER, D_MODEL);

        // causal conv + SiLU
        conv_silu_kernel<<<(MTOT * D_INNER) / 256, 256, 0, stream>>>(
            xz, conv_w + (size_t)i * D_INNER * D_CONV, conv_b + (size_t)i * D_INNER,
            xi_f32, xi_bf);

        // dbc = xi @ xproj^T : M=4096, N=64, K=1024
        wmma_gemm_bf16<<<dim3(1, MTOT / 128), 128, 0, stream>>>(
            xi_bf, D_INNER,
            w_xp_bf + (size_t)i * 64 * D_INNER, D_INNER,
            dbc, 64, MTOT, 64, D_INNER);

        castN(dbc, dbc_bf, MTOT * 64);

        // dtproj = dbc[:, :32] @ dt_w^T : M=4096, N=1024, K=32 (lda=64)
        wmma_gemm_bf16<<<dim3(D_INNER / 64, MTOT / 128), 128, 0, stream>>>(
            dbc_bf, 64,
            w_dt_bf + (size_t)i * D_INNER * DT_RANK, DT_RANK,
            dtproj, D_INNER, MTOT, D_INNER, DT_RANK);

        // selective scan (fused softplus / gate / D-skip / bf16 quant)
        scan_kernel<<<dim3(D_INNER / 256, BATCH), 256, 0, stream>>>(
            dtproj, dt_b + (size_t)i * D_INNER, dbc, xi_f32,
            A_log + (size_t)i * D_INNER * D_STATE, Dparam + (size_t)i * D_INNER,
            xz, y_bf);

        // h = y @ out_w^T : M=4096, N=512, K=1024
        wmma_gemm_bf16<<<dim3(D_MODEL / 64, MTOT / 128), 128, 0, stream>>>(
            y_bf, D_INNER,
            w_out_bf + (size_t)i * D_MODEL * D_INNER, D_INNER,
            h_f32, D_MODEL, MTOT, D_MODEL, D_INNER);

        castN(h_f32, h_bf, MTOT * D_MODEL);
    }

    // -------- head --------
    head_kernel<<<1, 64, 0, stream>>>(h_f32, head_w, head_b, out);
}
